// MultiheadAttnBlock_14456859918710
// MI455X (gfx1250) — matrix operand order repaired and checked by reading, not compiled, not run on hardware
//
#include <hip/hip_runtime.h>

// MI455X / gfx1250, wave32. Attention block:
//   GroupNorm -> QKV GEMM (WMMA f16) -> flash-style attention (WMMA f16,
//   async global->LDS double-buffered K/V staging) -> out projection + residual.

typedef _Float16 v16h __attribute__((ext_vector_type(16)));
typedef _Float16 v8h  __attribute__((ext_vector_type(8)));
typedef float    v8f  __attribute__((ext_vector_type(8)));
typedef float    v4f  __attribute__((ext_vector_type(4)));

static constexpr int Nn    = 16;    // B*T
static constexpr int Cc    = 512;
static constexpr int Ss    = 1024;  // H*W
static constexpr int HEADS = 8;
static constexpr int HD    = 64;
static constexpr int SWZ_XOR16 = 0x401f;  // group-of-32, xor 0x10, and 0x1f

// LDS tile geometry for attention: rows padded 64 -> 72 halves (144B = 36 banks
// per row step -> conflict-free 16-lane fragment reads, 16B aligned).
static constexpr int LROW  = 72;
static constexpr int KBYTES = 2 * 64 * LROW * 2;   // two K buffers
static constexpr int SMEM_BYTES = 2 * KBYTES;      // + two V buffers

__device__ __forceinline__ v8f wmma16(v16h a, v16h b, v8f c) {
  return __builtin_amdgcn_wmma_f32_16x16x32_f16(false, a, false, b, (short)0, c,
                                                false, false);
}
template <int IMM> __device__ __forceinline__ float swzf(float x) {
  union { float f; int i; } c; c.f = x;
  c.i = __builtin_amdgcn_ds_swizzle(c.i, IMM);
  return c.f;
}
template <int IMM> __device__ __forceinline__ unsigned swzu(unsigned x) {
  return (unsigned)__builtin_amdgcn_ds_swizzle((int)x, IMM);
}
__device__ __forceinline__ v16h cat8(v8h lo, v8h hi) {
  return __builtin_shufflevector(lo, hi, 0,1,2,3,4,5,6,7,8,9,10,11,12,13,14,15);
}
__device__ __forceinline__ unsigned pack2h(float a, float b) {
  union { _Float16 h[2]; unsigned u; } c;
  c.h[0] = (_Float16)a; c.h[1] = (_Float16)b;
  return c.u;
}
__device__ __forceinline__ v8f v8f_zero() {
  v8f z = {0.f,0.f,0.f,0.f,0.f,0.f,0.f,0.f};
  return z;
}
// gfx1250 async DMA: global -> LDS, 16B per lane, tracked by ASYNCcnt.
__device__ __forceinline__ void async_b128(unsigned lds_off, unsigned voff,
                                           unsigned long long sbase) {
  asm volatile("global_load_async_to_lds_b128 %0, %1, %2"
               :: "v"(lds_off), "v"(voff), "s"(sbase) : "memory");
}

// ---------------- 1) weight transpose (f32 row-major -> f16 transposed) ------
__global__ void wtrans_kernel(const float* __restrict__ src, _Float16* __restrict__ dst,
                              int rows, int cols) {
  __shared__ float tile[32][33];
  int tx = threadIdx.x & 31, ty = threadIdx.x >> 5;    // 32x8
  int c0 = blockIdx.x * 32, r0 = blockIdx.y * 32;
  #pragma unroll
  for (int i = 0; i < 32; i += 8)
    tile[ty + i][tx] = src[(size_t)(r0 + ty + i) * cols + c0 + tx];
  __syncthreads();
  #pragma unroll
  for (int i = 0; i < 32; i += 8)
    dst[(size_t)(c0 + ty + i) * rows + r0 + tx] = (_Float16)tile[tx][ty + i];
}

// ---------------- 2) GroupNorm: (n,c,s) f32 -> hn (pixel, channel) f16 -------
__global__ void gn_kernel(const float* __restrict__ x,
                          const float* __restrict__ gsc,
                          const float* __restrict__ gbi,
                          _Float16* __restrict__ hn) {
  int blk = blockIdx.x;                 // n*32 + g
  int n = blk >> 5, g = blk & 31;
  const float* xg = x + ((size_t)(n * Cc + g * 16)) * Ss;   // 16 ch x 1024 px
  int tid = threadIdx.x;

  float sum = 0.f, sq = 0.f;
  #pragma unroll 4
  for (int i = 0; i < 64; i++) {
    float v = xg[i * 256 + tid];
    sum += v; sq += v * v;
  }
  __shared__ float sb[512];
  sb[tid] = sum; sb[256 + tid] = sq;
  __syncthreads();
  for (int off = 128; off > 0; off >>= 1) {
    if (tid < off) { sb[tid] += sb[tid + off]; sb[256 + tid] += sb[256 + tid + off]; }
    __syncthreads();
  }
  float mu   = sb[0] * (1.f / 16384.f);
  float var  = sb[256] * (1.f / 16384.f) - mu * mu;
  float rstd = rsqrtf(var + 1e-6f);

  int cp = tid & 7, pb = tid >> 3;               // 8 threads per pixel, 2 ch each
  float s0 = gsc[g * 16 + cp * 2],     b0 = gbi[g * 16 + cp * 2];
  float s1 = gsc[g * 16 + cp * 2 + 1], b1 = gbi[g * 16 + cp * 2 + 1];
  #pragma unroll 4
  for (int it = 0; it < 32; it++) {
    int p = it * 32 + pb;
    float v0 = xg[(cp * 2) * Ss + p];
    float v1 = xg[(cp * 2 + 1) * Ss + p];
    unsigned u = pack2h((v0 - mu) * rstd * s0 + b0, (v1 - mu) * rstd * s1 + b1);
    *(unsigned*)(hn + ((size_t)(n * Ss + p)) * Cc + g * 16 + cp * 2) = u;
  }
}

// ---------------- 3) QKV GEMM: hn(16384x512) x WqkvT(1536x512)^T -------------
// Wave tile: M=32 (2x16), N=64 (4x16). Q,K stored (n,h,s,d); V stored (n,h,d,s).
__global__ void __launch_bounds__(256)
qkv_gemm_kernel(const _Float16* __restrict__ hn,
                const _Float16* __restrict__ wT,   // (1536,512)
                const float* __restrict__ bias,
                _Float16* __restrict__ Qb,
                _Float16* __restrict__ Kb,
                _Float16* __restrict__ Vt) {
  int wid  = (blockIdx.x << 3) | (threadIdx.x >> 5);
  int lane = threadIdx.x & 31;
  int lg = lane >> 4, lm = lane & 15;
  int mt = wid & 511;            // 512 m-tiles of 32 pixels
  int nt = wid >> 9;             // 24 n-tiles of 64 features
  int m0 = mt * 32, n0 = nt * 64;

  v8f acc[2][4];
  #pragma unroll
  for (int a = 0; a < 2; a++)
    #pragma unroll
    for (int b = 0; b < 4; b++) acc[a][b] = v8f_zero();

  for (int k0 = 0; k0 < Cc; k0 += 32) {
    v16h af[2];
    #pragma unroll
    for (int mi = 0; mi < 2; mi++) {
      const _Float16* p = hn + (size_t)(m0 + mi * 16 + lm) * Cc + k0 + lg * 8;
      af[mi] = cat8(*(const v8h*)p, *(const v8h*)(p + 16));
    }
    #pragma unroll
    for (int ni = 0; ni < 4; ni++) {
      const _Float16* p = wT + (size_t)(n0 + ni * 16 + lm) * Cc + k0 + lg * 8;
      v16h bf = cat8(*(const v8h*)p, *(const v8h*)(p + 16));
      acc[0][ni] = wmma16(af[0], bf, acc[0][ni]);
      acc[1][ni] = wmma16(af[1], bf, acc[1][ni]);
    }
  }

  #pragma unroll
  for (int ni = 0; ni < 4; ni++) {
    int f = n0 + ni * 16 + lm;           // 0..1535 (segment uniform per tile)
    float bv = bias[f];
    #pragma unroll
    for (int mi = 0; mi < 2; mi++) {
      int pix0 = m0 + mi * 16 + lg * 8;  // 8 consecutive pixels (j=0..7)
      int n = pix0 >> 10, sB = pix0 & 1023;
      if (f < 1024) {                    // Q or K: (n,h,s,d) row-major
        _Float16* base = (f < 512) ? Qb : Kb;
        int f2 = f & 511, h = f2 >> 6, d = f2 & 63;
        _Float16* dst = base + (((size_t)(n * HEADS + h) * Ss) + sB) * HD + d;
        #pragma unroll
        for (int j = 0; j < 8; j++)
          dst[(size_t)j * HD] = (_Float16)(acc[mi][ni][j] + bv);
      } else {                           // V transposed: (n,h,d,s)
        int f2 = f - 1024, h = f2 >> 6, d = f2 & 63;
        v8h o;
        #pragma unroll
        for (int j = 0; j < 8; j++) o[j] = (_Float16)(acc[mi][ni][j] + bv);
        *(v8h*)(Vt + (((size_t)(n * HEADS + h) * HD) + d) * Ss + sB) = o;
      }
    }
  }
}

// ---------------- 4) attention -----------------------------------------------
// Workgroup = 8 waves sharing one (n,h); each wave owns a 16-query tile.
// K/V 64-key tiles are DMA'd global->LDS (async, double-buffered) and consumed
// 8x from LDS. Transposed-logits flash attention: L^T = K * Q^T, per-lane
// softmax rows, P^T built with one xor-16 swizzle, O^T accumulated (d x s).
__global__ void __launch_bounds__(256)
attn_kernel(const _Float16* __restrict__ Qb,
            const _Float16* __restrict__ Kb,
            const _Float16* __restrict__ Vt,
            _Float16* __restrict__ Ob) {
  extern __shared__ _Float16 smem[];           // [2][64][LROW] K, then V
  _Float16* Ksh = smem;
  _Float16* Vsh = smem + 2 * 64 * LROW;

  int tid  = threadIdx.x;
  int wave = tid >> 5, lane = tid & 31;
  int lg = lane >> 4, lm = lane & 15;
  int nh = blockIdx.x >> 3;                    // (n*8+h)
  int qc = blockIdx.x & 7;                     // query chunk of 128
  const _Float16* q  = Qb + (size_t)nh * Ss * HD;
  unsigned long long kgB = (unsigned long long)(Kb + (size_t)nh * Ss * HD);
  unsigned long long vgB = (unsigned long long)(Vt + (size_t)nh * HD * Ss);
  int s0 = (qc * 8 + wave) * 16;

  // DMA lane roles: each thread moves 4x16B per tile (2 K chunks + 2 V chunks).
  int rowA = tid >> 3;                         // 0..31
  int col8 = tid & 7;                          // 8-half chunk within a row
  unsigned kL0 = (unsigned)((rowA * LROW + col8 * 8) * 2);
  unsigned kL1 = (unsigned)(((rowA + 32) * LROW + col8 * 8) * 2);
  unsigned vBase = (unsigned)(2 * 64 * LROW * 2);      // V region byte offset
  unsigned vL0 = vBase + kL0, vL1 = vBase + kL1;
  unsigned bufStride = (unsigned)(64 * LROW * 2);      // one buffer (9216B)

  // Q^T B-frags (d-ranges 0..31, 32..63), resident for the whole loop.
  v16h qf[2];
  #pragma unroll
  for (int dt = 0; dt < 2; dt++)
    qf[dt] = cat8(*(const v8h*)(q + (size_t)(s0 + lm) * HD + dt * 32 + lg * 8),
                  *(const v8h*)(q + (size_t)(s0 + lm) * HD + dt * 32 + lg * 8 + 16));

  const float scale = 0.125f * 1.4426950408889634f;    // (1/sqrt(64))*log2(e)
  float mrun = -3.0e38f, lrun = 0.f;
  v8f o[4];
  #pragma unroll
  for (int i = 0; i < 4; i++) o[i] = v8f_zero();

  // prologue: DMA tile 0 into buffer 0
  {
    unsigned kg0 = (unsigned)(((rowA) * 64 + col8 * 8) * 2);
    async_b128(kL0, kg0, kgB);
    async_b128(kL1, kg0 + 32u * 64u * 2u, kgB);
    unsigned vg0 = (unsigned)((rowA * 1024 + col8 * 8) * 2);
    async_b128(vL0, vg0, vgB);
    async_b128(vL1, vg0 + 32u * 1024u * 2u, vgB);
  }

  int buf = 0;
  for (int it = 0; it < 16; it++) {
    int tb = it * 64;
    if (it + 1 < 16) {                         // prefetch next tile, other buffer
      int tb2 = tb + 64;
      unsigned bo = (buf ^ 1) * bufStride;
      unsigned kg0 = (unsigned)(((tb2 + rowA) * 64 + col8 * 8) * 2);
      async_b128(kL0 + bo, kg0, kgB);
      async_b128(kL1 + bo, kg0 + 32u * 64u * 2u, kgB);
      unsigned vg0 = (unsigned)((rowA * 1024 + tb2 + col8 * 8) * 2);
      async_b128(vL0 + bo, vg0, vgB);
      async_b128(vL1 + bo, vg0 + 32u * 1024u * 2u, vgB);
      asm volatile("s_wait_asynccnt 0x4" ::: "memory");  // tile `it` landed
    } else {
      asm volatile("s_wait_asynccnt 0x0" ::: "memory");
    }
    __syncthreads();

    const _Float16* kt = Ksh + (size_t)buf * 64 * LROW;
    const _Float16* vt = Vsh + (size_t)buf * 64 * LROW;

    // logits^T tiles: 4 x (16t x 16s), K-dim d=64 in two WMMA steps
    v8f Lt[4];
    #pragma unroll
    for (int tt = 0; tt < 4; tt++) {
      v8f a = v8f_zero();
      #pragma unroll
      for (int dt = 0; dt < 2; dt++) {
        const _Float16* p = kt + (size_t)(tt * 16 + lm) * LROW + dt * 32 + lg * 8;
        a = wmma16(cat8(*(const v8h*)p, *(const v8h*)(p + 16)), qf[dt], a);
      }
      Lt[tt] = a;
    }

    // online softmax over 64 keys; lane owns row s = s0+lm
    float mloc = -3.0e38f;
    #pragma unroll
    for (int tt = 0; tt < 4; tt++)
      #pragma unroll
      for (int j = 0; j < 8; j++) mloc = fmaxf(mloc, Lt[tt][j]);
    mloc = fmaxf(mloc, swzf<SWZ_XOR16>(mloc));
    float mnew  = fmaxf(mrun, mloc);
    float alpha = __builtin_exp2f((mrun - mnew) * scale);

    unsigned pk[4][4];
    float lloc = 0.f;
    #pragma unroll
    for (int tt = 0; tt < 4; tt++)
      #pragma unroll
      for (int j = 0; j < 8; j += 2) {
        float p0 = __builtin_exp2f((Lt[tt][j]     - mnew) * scale);
        float p1 = __builtin_exp2f((Lt[tt][j + 1] - mnew) * scale);
        lloc += p0 + p1;
        pk[tt][j >> 1] = pack2h(p0, p1);
      }
    lrun = lrun * alpha + (lloc + swzf<SWZ_XOR16>(lloc));
    mrun = mnew;
    #pragma unroll
    for (int dt = 0; dt < 4; dt++)
      #pragma unroll
      for (int j = 0; j < 8; j++) o[dt][j] *= alpha;

    // P^T B-frags (two 32t x 16s) from logit D-frags via xor-16 exchange,
    // then O^T += Vt(16d x 32t) * P^T
    #pragma unroll
    for (int half = 0; half < 2; half++) {
      union { unsigned u[8]; v16h h; } pb;
      #pragma unroll
      for (int w = 0; w < 4; w++) {
        unsigned oth0 = swzu<SWZ_XOR16>(pk[2 * half][w]);
        unsigned oth1 = swzu<SWZ_XOR16>(pk[2 * half + 1][w]);
        pb.u[w]     = lg ? oth1 : pk[2 * half][w];
        pb.u[w + 4] = lg ? pk[2 * half + 1][w] : oth0;
      }
      #pragma unroll
      for (int dt = 0; dt < 4; dt++) {
        const _Float16* p = vt + (size_t)(dt * 16 + lm) * LROW + half * 32 + lg * 16;
        o[dt] = wmma16(cat8(*(const v8h*)p, *(const v8h*)(p + 8)), pb.h, o[dt]);
      }
    }
    __syncthreads();                           // buffer reuse guard
    buf ^= 1;
  }

  float inv = 1.f / lrun;
  int pix = (nh >> 3) * Ss + s0 + lm;
  int h   = nh & 7;
  #pragma unroll
  for (int dt = 0; dt < 4; dt++) {
    v8h ov;
    #pragma unroll
    for (int j = 0; j < 8; j++) ov[j] = (_Float16)(o[dt][j] * inv);
    *(v8h*)(Ob + (size_t)pix * Cc + h * HD + dt * 16 + lg * 8) = ov;
  }
}

// ---------------- 5) out projection + bias + residual (f32 out) --------------
__global__ void __launch_bounds__(256)
out_gemm_kernel(const _Float16* __restrict__ Ob,
                const _Float16* __restrict__ wT,   // (512,512)
                const float* __restrict__ bias,
                const float* __restrict__ x,
                float* __restrict__ out) {
  int wid  = (blockIdx.x << 3) | (threadIdx.x >> 5);
  int lane = threadIdx.x & 31;
  int lg = lane >> 4, lm = lane & 15;
  int mt = wid & 511, nt = wid >> 9;       // nt 0..7
  int m0 = mt * 32, n0 = nt * 64;

  v8f acc[2][4];
  #pragma unroll
  for (int a = 0; a < 2; a++)
    #pragma unroll
    for (int b = 0; b < 4; b++) acc[a][b] = v8f_zero();

  for (int k0 = 0; k0 < Cc; k0 += 32) {
    v16h af[2];
    #pragma unroll
    for (int mi = 0; mi < 2; mi++) {
      const _Float16* p = Ob + (size_t)(m0 + mi * 16 + lm) * Cc + k0 + lg * 8;
      af[mi] = cat8(*(const v8h*)p, *(const v8h*)(p + 16));
    }
    #pragma unroll
    for (int ni = 0; ni < 4; ni++) {
      const _Float16* p = wT + (size_t)(n0 + ni * 16 + lm) * Cc + k0 + lg * 8;
      v16h bf = cat8(*(const v8h*)p, *(const v8h*)(p + 16));
      acc[0][ni] = wmma16(af[0], bf, acc[0][ni]);
      acc[1][ni] = wmma16(af[1], bf, acc[1][ni]);
    }
  }

  #pragma unroll
  for (int ni = 0; ni < 4; ni++) {
    int f = n0 + ni * 16 + lm;
    float bv = bias[f];
    #pragma unroll
    for (int mi = 0; mi < 2; mi++) {
      int pix0 = m0 + mi * 16 + lg * 8;
      int n = pix0 >> 10, sB = pix0 & 1023;
      size_t off = ((size_t)(n * Cc + f)) * Ss + sB;
      v4f x0 = *(const v4f*)(x + off);
      v4f x1 = *(const v4f*)(x + off + 4);
      v4f r0, r1;
      #pragma unroll
      for (int j = 0; j < 4; j++) {
        r0[j] = x0[j] + acc[mi][ni][j]     + bv;
        r1[j] = x1[j] + acc[mi][ni][j + 4] + bv;
      }
      *(v4f*)(out + off)     = r0;
      *(v4f*)(out + off + 4) = r1;
    }
  }
}

extern "C" void kernel_launch(void* const* d_in, const int* in_sizes, int n_in,
                              void* d_out, int out_size, void* d_ws, size_t ws_size,
                              hipStream_t stream) {
  const float* x     = (const float*)d_in[0];
  const float* gsc   = (const float*)d_in[1];
  const float* gbi   = (const float*)d_in[2];
  const float* wqkv  = (const float*)d_in[3];   // (512, 1536)
  const float* bqkv  = (const float*)d_in[4];   // (1536)
  const float* wout  = (const float*)d_in[5];   // (512, 512)
  const float* bout  = (const float*)d_in[6];   // (512)
  float* out = (float*)d_out;

  const size_t ACT = (size_t)Nn * Ss * Cc;      // 8,388,608 halves
  _Float16* hn    = (_Float16*)d_ws;
  _Float16* Qb    = hn + ACT;
  _Float16* Kb    = Qb + ACT;
  _Float16* Vt    = Kb + ACT;
  _Float16* Ob    = Vt + ACT;
  _Float16* WqkvT = Ob + ACT;                   // 1536*512
  _Float16* WoutT = WqkvT + (size_t)1536 * 512; // 512*512

  wtrans_kernel<<<dim3(1536 / 32, 512 / 32), 256, 0, stream>>>(wqkv, WqkvT, 512, 1536);
  wtrans_kernel<<<dim3(512 / 32, 512 / 32), 256, 0, stream>>>(wout, WoutT, 512, 512);
  gn_kernel<<<Nn * 32, 256, 0, stream>>>(x, gsc, gbi, hn);
  qkv_gemm_kernel<<<1536, 256, 0, stream>>>(hn, WqkvT, bqkv, Qb, Kb, Vt);
  attn_kernel<<<1024, 256, SMEM_BYTES, stream>>>(Qb, Kb, Vt, Ob);
  out_gemm_kernel<<<512, 256, 0, stream>>>(Ob, WoutT, bout, x, out);
  (void)in_sizes; (void)n_in; (void)out_size; (void)ws_size;
}
